// FineGrainedPipelinedMoELayer_69561290326224
// MI455X (gfx1250) — compile-verified, matrix operand order checked by reading, not run on hardware
//
#include <hip/hip_runtime.h>

#define N_TOK 32768
#define DIM   1024
#define OUTD  1024
#define NEXP  64
#define TM    128
#define MAXT  (N_TOK / TM + NEXP)   // 320 row tiles max

typedef __attribute__((ext_vector_type(16))) __bf16 v16bf;
typedef __attribute__((ext_vector_type(8)))  float  v8f;

// Load 8 consecutive f32 and place as bf16 (RNE) into fragment elements [o..o+7].
static __device__ __forceinline__ void ld8(const float* __restrict__ p, v16bf& f, int o) {
  float4 a = *(const float4*)p;
  float4 b = *(const float4*)(p + 4);
  f[o + 0] = (__bf16)a.x; f[o + 1] = (__bf16)a.y;
  f[o + 2] = (__bf16)a.z; f[o + 3] = (__bf16)a.w;
  f[o + 4] = (__bf16)b.x; f[o + 5] = (__bf16)b.y;
  f[o + 6] = (__bf16)b.z; f[o + 7] = (__bf16)b.w;
}

// Truncating f32 -> bf16: reinterpret the high 16 bits (lowers to d16/d16_hi b16 loads).
static __device__ __forceinline__ __bf16 hibf(const unsigned short* __restrict__ h,
                                              size_t elem) {
  return __builtin_bit_cast(__bf16, h[2 * elem + 1]);
}

// ---------------- Gate: logits via WMMA, then per-token argmax ----------------
__global__ __launch_bounds__(128) void gate_kernel(const float* __restrict__ x,
                                                   const float* __restrict__ gw,
                                                   const float* __restrict__ gb,
                                                   int* __restrict__ sel) {
  __shared__ float lg[16][NEXP + 4];
  const int t0   = blockIdx.x * 16;
  const int wv   = threadIdx.x >> 5;       // 4 waves -> experts wv*16..+15
  const int lane = threadIdx.x & 31;
  const int m    = lane & 15;
  const int half = lane >> 4;
  const float* ap = x  + (size_t)(t0 + m) * DIM;       // A: 16 tokens (row-major)
  const float* bp = gw + (size_t)(wv * 16 + m) * DIM;  // B: gate_w rows are K-contiguous
  v8f acc = {};
  for (int kk = 0; kk < DIM / 32; ++kk) {
    const int kA = kk * 32 + half * 8;     // 16-bit operand layout: runs [kA..+7],[kA+16..+23]
    v16bf a, b;
    ld8(ap + kA, a, 0);       ld8(ap + kA + 16, a, 8);
    ld8(bp + kA, b, 0);       ld8(bp + kA + 16, b, 8);
    acc = __builtin_amdgcn_wmma_f32_16x16x32_bf16(false, a, false, b, (short)0, acc,
                                                  false, false);
  }
#pragma unroll
  for (int v = 0; v < 8; ++v)              // C layout: VGPR v -> M = v (+8 for hi lanes)
    lg[v + half * 8][wv * 16 + m] = acc[v];
  __syncthreads();
  if (threadIdx.x < 16) {
    const int r = threadIdx.x;
    float best = -3.4e38f; int bi = 0;
    for (int e = 0; e < NEXP; ++e) {
      float v = lg[r][e] + gb[e];
      if (v > best) { best = v; bi = e; }  // strict '>' => first max (matches argmax)
    }
    sel[t0 + r] = bi;
  }
}

// ---------------- Routing: histogram / scan+tile plan / scatter ----------------
__global__ void hist_kernel(const int* __restrict__ sel, int* __restrict__ counts) {
  int t = blockIdx.x * blockDim.x + threadIdx.x;
  if (t < N_TOK) atomicAdd(&counts[sel[t]], 1);
}

__global__ void plan_kernel(const int* __restrict__ counts, int* __restrict__ offsets,
                            int* __restrict__ cursor, int* __restrict__ tileExpert,
                            int* __restrict__ tileBase, int* __restrict__ numTiles) {
  if (threadIdx.x == 0 && blockIdx.x == 0) {
    int off = 0, nt = 0;
    for (int e = 0; e < NEXP; ++e) {
      offsets[e] = off; cursor[e] = off;
      for (int r = 0; r < counts[e]; r += TM) {
        if (nt < MAXT) { tileExpert[nt] = e; tileBase[nt] = off + r; ++nt; }
      }
      off += counts[e];
    }
    *numTiles = nt;
  }
}

__global__ void scatter_kernel(const int* __restrict__ sel, int* __restrict__ cursor,
                               int* __restrict__ sorted) {
  int t = blockIdx.x * blockDim.x + threadIdx.x;
  if (t < N_TOK) {
    int pos = atomicAdd(&cursor[sel[t]], 1);
    sorted[pos] = t;
  }
}

// ---------------- Grouped expert GEMM (WMMA) with scatter-on-store ----------------
// 256 threads = 8 waves arranged 2 row-groups x 4 col-groups; wave owns 64x32 C
// (4 row subtiles x 2 col subtiles, 8 v8f accumulators).
__global__ __launch_bounds__(256) void moe_gemm_kernel(
    const float* __restrict__ x, const float* __restrict__ ew,
    const float* __restrict__ eb, const int* __restrict__ sorted,
    const int* __restrict__ counts, const int* __restrict__ offsets,
    const int* __restrict__ tileExpert, const int* __restrict__ tileBase,
    const int* __restrict__ numTiles, float* __restrict__ out) {
  const int tile = blockIdx.y;
  if (tile >= *numTiles) return;           // uniform early-exit (before any barrier)
  __shared__ int stok[TM];
  const int e    = tileExpert[tile];
  const int base = tileBase[tile];
  const int rowsValid = min(TM, offsets[e] + counts[e] - base);
  if (threadIdx.x < TM)
    stok[threadIdx.x] = (threadIdx.x < (unsigned)rowsValid) ? sorted[base + threadIdx.x] : -1;
  __syncthreads();

  const int wv   = threadIdx.x >> 5;
  const int lane = threadIdx.x & 31;
  const int m    = lane & 15, half = lane >> 4;
  const int rg   = wv >> 2, cg = wv & 3;   // 2 row-groups x 4 col-groups
  const int r0   = rg * 64;

  int tok4[4];
  const float* ap[4];
#pragma unroll
  for (int s = 0; s < 4; ++s) {
    tok4[s] = stok[r0 + s * 16 + m];
    ap[s]   = x + (size_t)(tok4[s] < 0 ? 0 : tok4[s]) * DIM;  // clamp pads to token 0
  }
  const unsigned short* Wh = (const unsigned short*)(ew + (size_t)e * DIM * OUTD);
  const int ncolbase = blockIdx.x * 128 + cg * 32 + m;

  v8f acc[4][2] = {};
  for (int kk = 0; kk < DIM / 32; ++kk) {
    const int kA = kk * 32 + half * 8;
    v16bf a[4];
#pragma unroll
    for (int s = 0; s < 4; ++s) {
      ld8(ap[s] + kA, a[s], 0);
      ld8(ap[s] + kA + 16, a[s], 8);
    }
#pragma unroll
    for (int c = 0; c < 2; ++c) {
      const int n = ncolbase + c * 16;
      v16bf b;                             // B: K-strided, N = lane%16, truncated bf16
#pragma unroll
      for (int j = 0; j < 8; ++j)
        b[j] = hibf(Wh, (size_t)(kA + j) * OUTD + n);
#pragma unroll
      for (int j = 0; j < 8; ++j)
        b[8 + j] = hibf(Wh, (size_t)(kA + 16 + j) * OUTD + n);
#pragma unroll
      for (int s = 0; s < 4; ++s)
        acc[s][c] = __builtin_amdgcn_wmma_f32_16x16x32_bf16(false, a[s], false, b, (short)0,
                                                            acc[s][c], false, false);
    }
  }

#pragma unroll
  for (int c = 0; c < 2; ++c) {
    const int n = ncolbase + c * 16;
    const float bias = eb[(size_t)e * OUTD + n];
#pragma unroll
    for (int s = 0; s < 4; ++s) {
#pragma unroll
      for (int v = 0; v < 8; ++v) {
        const int row = r0 + s * 16 + v + half * 8;
        const int tok = stok[row];
        if (tok >= 0) out[(size_t)tok * OUTD + n] = acc[s][c][v] + bias;
      }
    }
  }
}

// ---------------- Host-side launch ----------------
extern "C" void kernel_launch(void* const* d_in, const int* in_sizes, int n_in,
                              void* d_out, int out_size, void* d_ws, size_t ws_size,
                              hipStream_t stream) {
  const float* x  = (const float*)d_in[0];
  const float* gw = (const float*)d_in[1];
  const float* gb = (const float*)d_in[2];
  const float* ew = (const float*)d_in[3];
  const float* eb = (const float*)d_in[4];
  float* out = (float*)d_out;

  int* Wp       = (int*)d_ws;
  int* sel      = Wp;
  int* sorted   = Wp + N_TOK;
  int* counts   = Wp + 2 * N_TOK;
  int* offsets  = counts + NEXP;
  int* cursor   = offsets + NEXP;
  int* numTiles = cursor + NEXP;
  int* tileExpert = numTiles + 1;
  int* tileBase   = tileExpert + MAXT;

  hipMemsetAsync(counts, 0, NEXP * sizeof(int), stream);
  gate_kernel<<<N_TOK / 16, 128, 0, stream>>>(x, gw, gb, sel);
  hist_kernel<<<N_TOK / 256, 256, 0, stream>>>(sel, counts);
  plan_kernel<<<1, 64, 0, stream>>>(counts, offsets, cursor, tileExpert, tileBase, numTiles);
  scatter_kernel<<<N_TOK / 256, 256, 0, stream>>>(sel, cursor, sorted);
  moe_gemm_kernel<<<dim3(OUTD / 128, MAXT), 256, 0, stream>>>(
      x, ew, eb, sorted, counts, offsets, tileExpert, tileBase, numTiles, out);
}
